// NostrSAGE_9363028705393
// MI455X (gfx1250) — matrix operand order repaired and checked by reading, not compiled, not run on hardware
//
#include <hip/hip_runtime.h>
#include <stdint.h>

// ---------- WMMA vector types (CDNA5 / gfx1250, wave32) ----------
typedef __attribute__((ext_vector_type(16))) __bf16          bf16x16;
typedef __attribute__((ext_vector_type(16))) unsigned short  u16x16;
typedef __attribute__((ext_vector_type(8)))  float           f32x8;

// 32-byte register tile: 16 bf16 elements, fillable via two b128 loads.
union AVec {
    u16x16  u;
    bf16x16 b;
    uint4   q[2];
};

// fp32 -> bf16 round-to-nearest-even (bit exact, no __bf16 arithmetic)
__device__ __forceinline__ unsigned short f2bf(float f) {
    unsigned u = __float_as_uint(f);
    u += 0x7FFFu + ((u >> 16) & 1u);
    return (unsigned short)(u >> 16);
}
__device__ __forceinline__ float bf2f(unsigned short h) {
    return __uint_as_float(((unsigned)h) << 16);
}

// ---------- degree count / reciprocal ----------
__global__ void k_count(const long long* __restrict__ edst, float* __restrict__ cnt, int nE) {
    int e = blockIdx.x * blockDim.x + threadIdx.x;
    if (e < nE) atomicAdd(&cnt[(int)edst[e]], 1.0f);
}

__global__ void k_inv(const float* __restrict__ cnt, float* __restrict__ inv, int n) {
    int i = blockIdx.x * blockDim.x + threadIdx.x;
    if (i < n) inv[i] = 1.0f / fmaxf(cnt[i], 1.0f);
}

// ---------- edge scatter: one wave per edge, float4 per lane, f32 atomics ----------
__global__ void k_scatter(const float* __restrict__ feat,
                          const long long* __restrict__ esrc,
                          const long long* __restrict__ edst,
                          float* __restrict__ agg, int nE) {
    const int lane = threadIdx.x & 31;
    const int e = blockIdx.x * (blockDim.x >> 5) + (threadIdx.x >> 5);
    if (e >= nE) return;
    const int s = (int)esrc[e];
    const int d = (int)edst[e];
    const float4 v = reinterpret_cast<const float4*>(feat + (size_t)s * 128)[lane];
    float* dp = agg + (size_t)d * 128 + lane * 4;
    atomicAdd(dp + 0, v.x);
    atomicAdd(dp + 1, v.y);
    atomicAdd(dp + 2, v.z);
    atomicAdd(dp + 3, v.w);
}

// ---------- one-shot fp32 -> bf16(hi) + bf16(residual) packing ----------
// 4 elements per thread: float4 in, two ushort4 out.
__global__ void k_pack(const float* __restrict__ in,
                       unsigned short* __restrict__ hi,
                       unsigned short* __restrict__ lo, int n4) {
    int i = blockIdx.x * blockDim.x + threadIdx.x;
    if (i >= n4) return;
    const float4 v = reinterpret_cast<const float4*>(in)[i];
    const float f[4] = {v.x, v.y, v.z, v.w};
    ushort4 h4, l4;
    unsigned short* hp = &h4.x;
    unsigned short* lp = &l4.x;
    #pragma unroll
    for (int e = 0; e < 4; ++e) {
        unsigned short h = f2bf(f[e]);
        hp[e] = h;
        lp[e] = f2bf(f[e] - bf2f(h));
    }
    reinterpret_cast<ushort4*>(hi)[i] = h4;
    reinterpret_cast<ushort4*>(lo)[i] = l4;
}

// Same, but computes mean = agg * inv_cnt[node] on the fly (row width fixed 128).
__global__ void k_pack_mean(const float* __restrict__ agg,
                            const float* __restrict__ invc,
                            unsigned short* __restrict__ hi,
                            unsigned short* __restrict__ lo, int n4) {
    int i = blockIdx.x * blockDim.x + threadIdx.x;
    if (i >= n4) return;
    const float inv = invc[(i * 4) >> 7];
    const float4 v = reinterpret_cast<const float4*>(agg)[i];
    const float f[4] = {v.x * inv, v.y * inv, v.z * inv, v.w * inv};
    ushort4 h4, l4;
    unsigned short* hp = &h4.x;
    unsigned short* lp = &l4.x;
    #pragma unroll
    for (int e = 0; e < 4; ++e) {
        unsigned short h = f2bf(f[e]);
        hp[e] = h;
        lp[e] = f2bf(f[e] - bf2f(h));
    }
    reinterpret_cast<ushort4*>(hi)[i] = h4;
    reinterpret_cast<ushort4*>(lo)[i] = l4;
}

// ---------- GEMM: out = mean@Wl^T + x@Wr^T + b [+ReLU], packed-bf16 WMMA ----------
// One wave = one 16(node) x 16(channel) tile. K = 128 in four 32-chunks.
// All operands are pre-split bf16 hi/lo planes, row-major [rows,128].
template <int OUTC, bool RELU>
__global__ void k_sage_gemm(const unsigned short* __restrict__ aMh,  // mean hi   [N,128]
                            const unsigned short* __restrict__ aMl,  // mean lo
                            const unsigned short* __restrict__ aXh,  // x    hi
                            const unsigned short* __restrict__ aXl,  // x    lo
                            const unsigned short* __restrict__ wLh,  // Wl hi [OUTC,128]
                            const unsigned short* __restrict__ wLl,
                            const unsigned short* __restrict__ wRh,  // Wr hi
                            const unsigned short* __restrict__ wRl,
                            const float* __restrict__ bias,
                            float* __restrict__ out, int nNodes) {
    const int lane   = threadIdx.x & 31;
    const int wave   = threadIdx.x >> 5;
    const int tile_m = blockIdx.x;
    const int tile_n = blockIdx.y * 4 + wave;
    if (tile_n * 16 >= OUTC) return;

    const int m    = lane & 15;
    const int half = lane >> 4;

    int node = tile_m * 16 + m;
    if (node >= nNodes) node = nNodes - 1;       // clamp; OOB rows never stored

    const size_t aOff = (size_t)node * 128;
    const int    orow = tile_n * 16 + m;         // weight row / out channel of lane
    const size_t bOff = (size_t)orow * 128;

    f32x8 acc = {};

    #pragma unroll
    for (int kc = 0; kc < 4; ++kc) {
        AVec vMh, vMl, vXh, vXl, bLh, bLl, bRh, bRl;

        // A (16x32 operand, contiguous k order used here, same as B): elem e -> K = half*16 + e
        // => two 8-element runs, each one b128 load.
        const int k0 = kc * 32 + half * 16;
        const int k1 = k0 + 8;
        vMh.q[0] = *reinterpret_cast<const uint4*>(aMh + aOff + k0);
        vMh.q[1] = *reinterpret_cast<const uint4*>(aMh + aOff + k1);
        vMl.q[0] = *reinterpret_cast<const uint4*>(aMl + aOff + k0);
        vMl.q[1] = *reinterpret_cast<const uint4*>(aMl + aOff + k1);
        vXh.q[0] = *reinterpret_cast<const uint4*>(aXh + aOff + k0);
        vXh.q[1] = *reinterpret_cast<const uint4*>(aXh + aOff + k1);
        vXl.q[0] = *reinterpret_cast<const uint4*>(aXl + aOff + k0);
        vXl.q[1] = *reinterpret_cast<const uint4*>(aXl + aOff + k1);

        // B (32x16, k order matches A): lane column n = lane&15, elem e -> K = half*16 + e
        // => 16 consecutive u16 per lane = two b128 loads.
        const int kb = kc * 32 + half * 16;
        bLh.q[0] = *reinterpret_cast<const uint4*>(wLh + bOff + kb);
        bLh.q[1] = *reinterpret_cast<const uint4*>(wLh + bOff + kb + 8);
        bLl.q[0] = *reinterpret_cast<const uint4*>(wLl + bOff + kb);
        bLl.q[1] = *reinterpret_cast<const uint4*>(wLl + bOff + kb + 8);
        bRh.q[0] = *reinterpret_cast<const uint4*>(wRh + bOff + kb);
        bRh.q[1] = *reinterpret_cast<const uint4*>(wRh + bOff + kb + 8);
        bRl.q[0] = *reinterpret_cast<const uint4*>(wRl + bOff + kb);
        bRl.q[1] = *reinterpret_cast<const uint4*>(wRl + bOff + kb + 8);

        // Split-precision: hi*hi + lo*hi + hi*lo, f32 accumulate
        acc = __builtin_amdgcn_wmma_f32_16x16x32_bf16(false, vMh.b, false, bLh.b, (short)0, acc, false, false);
        acc = __builtin_amdgcn_wmma_f32_16x16x32_bf16(false, vMl.b, false, bLh.b, (short)0, acc, false, false);
        acc = __builtin_amdgcn_wmma_f32_16x16x32_bf16(false, vMh.b, false, bLl.b, (short)0, acc, false, false);
        acc = __builtin_amdgcn_wmma_f32_16x16x32_bf16(false, vXh.b, false, bRh.b, (short)0, acc, false, false);
        acc = __builtin_amdgcn_wmma_f32_16x16x32_bf16(false, vXl.b, false, bRh.b, (short)0, acc, false, false);
        acc = __builtin_amdgcn_wmma_f32_16x16x32_bf16(false, vXh.b, false, bRl.b, (short)0, acc, false, false);
    }

    // C/D layout: lane column n = lane&15, VGPR r -> row m = r + 8*half
    const float bv = bias[orow];
    #pragma unroll
    for (int r = 0; r < 8; ++r) {
        const int nodeOut = tile_m * 16 + r + half * 8;
        if (nodeOut < nNodes) {
            float v = acc[r] + bv;
            if (RELU) v = fmaxf(v, 0.0f);
            out[(size_t)nodeOut * OUTC + orow] = v;
        }
    }
}

extern "C" void kernel_launch(void* const* d_in, const int* in_sizes, int n_in,
                              void* d_out, int out_size, void* d_ws, size_t ws_size,
                              hipStream_t stream) {
    const float*     x    = (const float*)d_in[0];
    const long long* eidx = (const long long*)d_in[1];   // int64 [2,E]
    const float*     W1l  = (const float*)d_in[2];
    const float*     b1l  = (const float*)d_in[3];
    const float*     W1r  = (const float*)d_in[4];
    const float*     W2l  = (const float*)d_in[5];
    const float*     b2l  = (const float*)d_in[6];
    const float*     W2r  = (const float*)d_in[7];
    float*           out  = (float*)d_out;

    const int N = in_sizes[0] / 128;
    const int E = in_sizes[1] / 2;
    const long long* esrc = eidx;
    const long long* edst = eidx + E;

    // ---- workspace carve-up (256B aligned blocks) ----
    char* p = (char*)d_ws;
    auto alloc = [&](size_t bytes) -> void* {
        void* r = p;
        p += (bytes + 255) & ~(size_t)255;
        return r;
    };
    const size_t NF = (size_t)N * 128;
    float* cnt  = (float*)alloc(sizeof(float) * N);
    float* invc = (float*)alloc(sizeof(float) * N);
    float* agg  = (float*)alloc(sizeof(float) * NF);
    float* h    = (float*)alloc(sizeof(float) * NF);
    unsigned short* pAmHi = (unsigned short*)alloc(2 * NF);
    unsigned short* pAmLo = (unsigned short*)alloc(2 * NF);
    unsigned short* pAxHi = (unsigned short*)alloc(2 * NF);
    unsigned short* pAxLo = (unsigned short*)alloc(2 * NF);
    unsigned short* w1lHi = (unsigned short*)alloc(2 * 128 * 128);
    unsigned short* w1lLo = (unsigned short*)alloc(2 * 128 * 128);
    unsigned short* w1rHi = (unsigned short*)alloc(2 * 128 * 128);
    unsigned short* w1rLo = (unsigned short*)alloc(2 * 128 * 128);
    unsigned short* w2lHi = (unsigned short*)alloc(2 * 64 * 128);
    unsigned short* w2lLo = (unsigned short*)alloc(2 * 64 * 128);
    unsigned short* w2rHi = (unsigned short*)alloc(2 * 64 * 128);
    unsigned short* w2rLo = (unsigned short*)alloc(2 * 64 * 128);

    hipMemsetAsync(cnt, 0, sizeof(float) * (size_t)N, stream);
    hipMemsetAsync(agg, 0, sizeof(float) * NF, stream);

    k_count<<<(E + 255) / 256, 256, 0, stream>>>(edst, cnt, E);
    k_inv<<<(N + 255) / 256, 256, 0, stream>>>(cnt, invc, N);

    // ---- pack weights once ----
    const int wq1 = (128 * 128) / 4, wq2 = (64 * 128) / 4;
    k_pack<<<(wq1 + 255) / 256, 256, 0, stream>>>(W1l, w1lHi, w1lLo, wq1);
    k_pack<<<(wq1 + 255) / 256, 256, 0, stream>>>(W1r, w1rHi, w1rLo, wq1);
    k_pack<<<(wq2 + 255) / 256, 256, 0, stream>>>(W2l, w2lHi, w2lLo, wq2);
    k_pack<<<(wq2 + 255) / 256, 256, 0, stream>>>(W2r, w2rHi, w2rLo, wq2);

    const int nq = (int)(NF / 4);
    const int tilesM = (N + 15) / 16;

    // ---- Layer 1 ----
    k_scatter<<<(E + 7) / 8, 256, 0, stream>>>(x, esrc, edst, agg, E);
    k_pack<<<(nq + 255) / 256, 256, 0, stream>>>(x, pAxHi, pAxLo, nq);
    k_pack_mean<<<(nq + 255) / 256, 256, 0, stream>>>(agg, invc, pAmHi, pAmLo, nq);
    dim3 g1(tilesM, 2);                       // 2 blocks.y * 4 waves -> 128 out channels
    k_sage_gemm<128, true><<<g1, 128, 0, stream>>>(pAmHi, pAmLo, pAxHi, pAxLo,
                                                   w1lHi, w1lLo, w1rHi, w1rLo,
                                                   b1l, h, N);

    // ---- Layer 2 ----
    hipMemsetAsync(agg, 0, sizeof(float) * NF, stream);
    k_scatter<<<(E + 7) / 8, 256, 0, stream>>>(h, esrc, edst, agg, E);
    k_pack<<<(nq + 255) / 256, 256, 0, stream>>>(h, pAxHi, pAxLo, nq);
    k_pack_mean<<<(nq + 255) / 256, 256, 0, stream>>>(agg, invc, pAmHi, pAmLo, nq);
    dim3 g2(tilesM, 1);                       // 4 waves -> 64 out channels
    k_sage_gemm<64, false><<<g2, 128, 0, stream>>>(pAmHi, pAmLo, pAxHi, pAxLo,
                                                   w2lHi, w2lLo, w2rHi, w2rLo,
                                                   b2l, out, N);
}
